// MEGNetGraphConv_90864328114391
// MI455X (gfx1250) — matrix operand order repaired and checked by reading, not compiled, not run on hardware
//
#include <hip/hip_runtime.h>
#include <math.h>

#define N_NODES  50000
#define N_EDGES  800000
#define N_GRAPHS 8
#define DD 32
#define HH 64
#define BSTR 40   // halves per (chunk,column) of swizzled B; 80B stride, 16B aligned, bank-conflict-free

typedef __attribute__((ext_vector_type(16))) _Float16 v16h;
typedef __attribute__((ext_vector_type(8)))  _Float16 v8h;
typedef __attribute__((ext_vector_type(8)))  float    v8f;
typedef _Float16 h16;

__device__ __forceinline__ float sp(float x) {
    // softplus(x) = log(1+exp(x)), numerically stable
    return fmaxf(x, 0.f) + log1pf(expf(-fabsf(x)));
}

// Quarter swizzle inside a 32-half K-chunk: logical k -> physical pos.
// Quarters 1 (k=8..15) and 2 (k=16..23) are swapped so that each lane-half of an
// A fragment is 16 contiguous halves (2 x ds_load_b128).
__device__ __forceinline__ int swzc(int c) {
    const int q = c >> 3;
    const int pq = (q == 1) ? 2 : ((q == 2) ? 1 : q);
    return pq * 8 + (c & 7);
}

// A fragment (16x32 f16, MxK) from a swizzled row-major LDS tile, leading dim ld halves.
__device__ __forceinline__ v16h load_A_sw(const h16* __restrict__ X, int ld, int kc, int lane) {
    const int m = lane & 15, hi = lane >> 4;
    const h16* p = X + m * ld + kc * 32 + hi * 16;
    v8h lo = *(const v8h*)p;
    v8h hh = *(const v8h*)(p + 8);
    return __builtin_shufflevector(lo, hh, 0, 1, 2, 3, 4, 5, 6, 7, 8, 9, 10, 11, 12, 13, 14, 15);
}

// B fragment (32x16 f16, KxN) from K-major swizzled weights:
// Wsw[(kc*ncols + n)*BSTR + k], k = 0..31 within chunk kc.
__device__ __forceinline__ v16h load_B_sw(const h16* __restrict__ Wsw, int ncols, int kc, int nb, int lane) {
    const int n = nb + (lane & 15);
    const h16* p = Wsw + (size_t)(kc * ncols + n) * BSTR + ((lane >> 4) * 16);
    v8h lo = *(const v8h*)p;
    v8h hh = *(const v8h*)(p + 8);
    return __builtin_shufflevector(lo, hh, 0, 1, 2, 3, 4, 5, 6, 7, 8, 9, 10, 11, 12, 13, 14, 15);
}

// Convert one 32-float segment to f16 and store it as one swizzled 32-half chunk
// (4 x 16-byte LDS stores, quarter-permuted).
__device__ __forceinline__ void cvt_store_chunk(h16* __restrict__ chunk,
                                                const float* __restrict__ src, float scale) {
    const float4* s4 = (const float4*)src;
#pragma unroll
    for (int q = 0; q < 4; ++q) {
        const float4 a = s4[2 * q], b = s4[2 * q + 1];
        v8h t;
        t[0] = (h16)(a.x * scale); t[1] = (h16)(a.y * scale);
        t[2] = (h16)(a.z * scale); t[3] = (h16)(a.w * scale);
        t[4] = (h16)(b.x * scale); t[5] = (h16)(b.y * scale);
        t[6] = (h16)(b.z * scale); t[7] = (h16)(b.w * scale);
        const int pq = (q == 1) ? 2 : ((q == 2) ? 1 : q);
        *(v8h*)(chunk + pq * 8) = t;
    }
}

__device__ __forceinline__ void zero_chunk(h16* __restrict__ chunk) {
    v8h z;
#pragma unroll
    for (int j = 0; j < 8; ++j) z[j] = (h16)0.f;
#pragma unroll
    for (int q = 0; q < 4; ++q) *(v8h*)(chunk + q * 8) = z;
}

// Cooperative load of row-major f32 weights W[K][N] into K-major swizzled f16 LDS.
__device__ __forceinline__ void load_weights_sw(h16* __restrict__ Wsw, const float* __restrict__ W,
                                                int K, int N, int tid, int nthr) {
    for (int i = tid; i < K * N; i += nthr) {
        const int kg = i / N, n = i - kg * N;
        Wsw[(size_t)((kg >> 5) * N + n) * BSTR + swzc(kg & 31)] = (h16)W[i];
    }
}

#define WMMA_F16(a, b, c) \
    __builtin_amdgcn_wmma_f32_16x16x32_f16(false, (a), false, (b), (short)0, (c), false, false)

// ------------------------------------------------------------------ edge stage
__global__ __launch_bounds__(128) void megnet_edge_kernel(
    const float* __restrict__ edge_feat, const float* __restrict__ node_feat,
    const float* __restrict__ graph_attr, const int* __restrict__ src,
    const int* __restrict__ dst, const int* __restrict__ ngid,
    const float* __restrict__ W0, const float* __restrict__ b0,
    const float* __restrict__ W1, const float* __restrict__ b1,
    float* __restrict__ e_new, float* __restrict__ e_sum, float* __restrict__ deg,
    float* __restrict__ ue, float* __restrict__ ne)
{
    __shared__ h16 W0s[4 * 64 * BSTR];   // 128x64 weights, K-major swizzled
    __shared__ h16 W1s[2 * 32 * BSTR];   // 64x32
    __shared__ h16 Xt[4][16 * 136];      // 16 x 128 (+pad) swizzled input tile per wave
    __shared__ h16 Ht[4][16 * 72];       // 16 x 64 (+pad) swizzled hidden tile per wave
    __shared__ int rsrc[4][16], rdst[4][16], rgid[4][16];
    __shared__ float lue[N_GRAPHS * DD];
    __shared__ float lne[N_GRAPHS];

    const int tid = threadIdx.x, lane = tid & 31, w = tid >> 5;
    const int e0 = (blockIdx.x * 4 + w) * 16;

    load_weights_sw(W0s, W0, 128, 64, tid, 128);
    load_weights_sw(W1s, W1, 64, 32, tid, 128);
    for (int i = tid; i < N_GRAPHS * DD; i += 128) lue[i] = 0.f;
    if (tid < N_GRAPHS) lne[tid] = 0.f;

    if (lane < 16) {
        const int e = e0 + lane;
        const int s = src[e], d = dst[e];
        const int g = ngid[s];
        rsrc[w][lane] = s; rdst[w][lane] = d; rgid[w][lane] = g;
        atomicAdd(&deg[d], 1.f);
        atomicAdd(&lne[g], 1.f);
    }
    __syncthreads();

    // gather X = [v_src | v_dst | e | u_src] (two lanes per row, two 32-col chunks each)
    {
        const int r = lane >> 1;
        h16* xr = &Xt[w][r * 136];
        if ((lane & 1) == 0) {
            cvt_store_chunk(xr,      node_feat + (size_t)rsrc[w][r] * DD, 1.f);
            cvt_store_chunk(xr + 32, node_feat + (size_t)rdst[w][r] * DD, 1.f);
        } else {
            cvt_store_chunk(xr + 64, edge_feat + (size_t)(e0 + r) * DD, 1.f);
            cvt_store_chunk(xr + 96, graph_attr + (size_t)rgid[w][r] * DD, 1.f);
        }
    }
    __syncthreads();

    const int n = lane & 15, hi = lane >> 4;

    // layer 1: [16x128] @ [128x64] + b0, softplus
#pragma unroll
    for (int nt = 0; nt < 4; ++nt) {
        const float bias = b0[nt * 16 + n];
        v8f c;
#pragma unroll
        for (int r = 0; r < 8; ++r) c[r] = bias;
#pragma unroll
        for (int kc = 0; kc < 4; ++kc) {
            v16h a  = load_A_sw(Xt[w], 136, kc, lane);
            v16h bf = load_B_sw(W0s, 64, kc, nt * 16, lane);
            c = WMMA_F16(a, bf, c);
        }
#pragma unroll
        for (int r = 0; r < 8; ++r) {
            const int m = r + 8 * hi;
            const int col = nt * 16 + n;
            Ht[w][m * 72 + (col >> 5) * 32 + swzc(col & 31)] = (h16)sp(c[r]);
        }
    }
    __syncthreads();

    // layer 2: [16x64] @ [64x32] + b1, softplus -> e_new + scatter adds
#pragma unroll
    for (int nt = 0; nt < 2; ++nt) {
        const float bias = b1[nt * 16 + n];
        v8f c;
#pragma unroll
        for (int r = 0; r < 8; ++r) c[r] = bias;
#pragma unroll
        for (int kc = 0; kc < 2; ++kc) {
            v16h a  = load_A_sw(Ht[w], 72, kc, lane);
            v16h bf = load_B_sw(W1s, 32, kc, nt * 16, lane);
            c = WMMA_F16(a, bf, c);
        }
#pragma unroll
        for (int r = 0; r < 8; ++r) {
            const int m = r + 8 * hi;
            const int col = nt * 16 + n;
            const float v = sp(c[r]);
            e_new[(size_t)(e0 + m) * DD + col] = v;
            atomicAdd(&e_sum[(size_t)rdst[w][m] * DD + col], v);
            atomicAdd(&lue[rgid[w][m] * DD + col], v);
        }
    }
    __syncthreads();

    for (int i = tid; i < N_GRAPHS * DD; i += 128) atomicAdd(&ue[i], lue[i]);
    if (tid < N_GRAPHS) atomicAdd(&ne[tid], lne[tid]);
}

// ------------------------------------------------------------------ node stage
__global__ __launch_bounds__(128) void megnet_node_kernel(
    const float* __restrict__ node_feat, const float* __restrict__ graph_attr,
    const int* __restrict__ ngid,
    const float* __restrict__ W0, const float* __restrict__ b0,
    const float* __restrict__ W1, const float* __restrict__ b1,
    const float* __restrict__ e_sum, const float* __restrict__ deg,
    float* __restrict__ v_new, float* __restrict__ uv, float* __restrict__ nv)
{
    __shared__ h16 W0s[3 * 64 * BSTR];   // 96x64
    __shared__ h16 W1s[2 * 32 * BSTR];   // 64x32
    __shared__ h16 Xt[4][16 * 104];      // 16 x 96 (+pad)
    __shared__ h16 Ht[4][16 * 72];
    __shared__ int rgid[4][16];
    __shared__ float luv[N_GRAPHS * DD];
    __shared__ float lnv[N_GRAPHS];

    const int tid = threadIdx.x, lane = tid & 31, w = tid >> 5;
    const int n0 = (blockIdx.x * 4 + w) * 16;

    load_weights_sw(W0s, W0, 96, 64, tid, 128);
    load_weights_sw(W1s, W1, 64, 32, tid, 128);
    for (int i = tid; i < N_GRAPHS * DD; i += 128) luv[i] = 0.f;
    if (tid < N_GRAPHS) lnv[tid] = 0.f;

    if (lane < 16) {
        const int node = n0 + lane;
        int g = 0;
        if (node < N_NODES) {
            g = ngid[node];
            atomicAdd(&lnv[g], 1.f);
        }
        rgid[w][lane] = g;
    }
    __syncthreads();

    // gather X = [v | mean_e | u]
    {
        const int r = lane >> 1;
        const int node = n0 + r;
        h16* xr = &Xt[w][r * 104];
        if (node < N_NODES) {
            if ((lane & 1) == 0) {
                cvt_store_chunk(xr, node_feat + (size_t)node * DD, 1.f);
                const float idg = 1.f / fmaxf(deg[node], 1.f);
                cvt_store_chunk(xr + 32, e_sum + (size_t)node * DD, idg);
            } else {
                cvt_store_chunk(xr + 64, graph_attr + (size_t)rgid[w][r] * DD, 1.f);
            }
        } else {
            if ((lane & 1) == 0) {
                zero_chunk(xr);
                zero_chunk(xr + 32);
            } else {
                zero_chunk(xr + 64);
            }
        }
    }
    __syncthreads();

    const int n = lane & 15, hi = lane >> 4;

    // layer 1: [16x96] @ [96x64]
#pragma unroll
    for (int nt = 0; nt < 4; ++nt) {
        const float bias = b0[nt * 16 + n];
        v8f c;
#pragma unroll
        for (int r = 0; r < 8; ++r) c[r] = bias;
#pragma unroll
        for (int kc = 0; kc < 3; ++kc) {
            v16h a  = load_A_sw(Xt[w], 104, kc, lane);
            v16h bf = load_B_sw(W0s, 64, kc, nt * 16, lane);
            c = WMMA_F16(a, bf, c);
        }
#pragma unroll
        for (int r = 0; r < 8; ++r) {
            const int m = r + 8 * hi;
            const int col = nt * 16 + n;
            Ht[w][m * 72 + (col >> 5) * 32 + swzc(col & 31)] = (h16)sp(c[r]);
        }
    }
    __syncthreads();

    // layer 2: [16x64] @ [64x32]
#pragma unroll
    for (int nt = 0; nt < 2; ++nt) {
        const float bias = b1[nt * 16 + n];
        v8f c;
#pragma unroll
        for (int r = 0; r < 8; ++r) c[r] = bias;
#pragma unroll
        for (int kc = 0; kc < 2; ++kc) {
            v16h a  = load_A_sw(Ht[w], 72, kc, lane);
            v16h bf = load_B_sw(W1s, 32, kc, nt * 16, lane);
            c = WMMA_F16(a, bf, c);
        }
#pragma unroll
        for (int r = 0; r < 8; ++r) {
            const int m = r + 8 * hi;
            const int node = n0 + m;
            if (node < N_NODES) {
                const int col = nt * 16 + n;
                const float v = sp(c[r]);
                v_new[(size_t)node * DD + col] = v;
                atomicAdd(&luv[rgid[w][m] * DD + col], v);
            }
        }
    }
    __syncthreads();

    for (int i = tid; i < N_GRAPHS * DD; i += 128) atomicAdd(&uv[i], luv[i]);
    if (tid < N_GRAPHS) atomicAdd(&nv[tid], lnv[tid]);
}

// ------------------------------------------------------------------ attr stage
__global__ __launch_bounds__(32) void megnet_attr_kernel(
    const float* __restrict__ graph_attr,
    const float* __restrict__ W0, const float* __restrict__ b0,
    const float* __restrict__ W1, const float* __restrict__ b1,
    const float* __restrict__ ue, const float* __restrict__ ne,
    const float* __restrict__ uv, const float* __restrict__ nv,
    float* __restrict__ u_new)
{
    __shared__ h16 W0s[3 * 64 * BSTR];
    __shared__ h16 W1s[2 * 32 * BSTR];
    __shared__ h16 Xt[16 * 104];
    __shared__ h16 Ht[16 * 72];

    const int lane = threadIdx.x;
    load_weights_sw(W0s, W0, 96, 64, lane, 32);
    load_weights_sw(W1s, W1, 64, 32, lane, 32);

    if (lane < 16) {
        h16* xr = &Xt[lane * 104];
        if (lane < N_GRAPHS) {
            const float ine = 1.f / fmaxf(ne[lane], 1.f);
            const float inv = 1.f / fmaxf(nv[lane], 1.f);
            cvt_store_chunk(xr,      graph_attr + (size_t)lane * DD, 1.f);
            cvt_store_chunk(xr + 32, ue + (size_t)lane * DD, ine);
            cvt_store_chunk(xr + 64, uv + (size_t)lane * DD, inv);
        } else {
            zero_chunk(xr);
            zero_chunk(xr + 32);
            zero_chunk(xr + 64);
        }
    }
    __syncthreads();

    const int n = lane & 15, hi = lane >> 4;

#pragma unroll
    for (int nt = 0; nt < 4; ++nt) {
        const float bias = b0[nt * 16 + n];
        v8f c;
#pragma unroll
        for (int r = 0; r < 8; ++r) c[r] = bias;
#pragma unroll
        for (int kc = 0; kc < 3; ++kc) {
            v16h a  = load_A_sw(Xt, 104, kc, lane);
            v16h bf = load_B_sw(W0s, 64, kc, nt * 16, lane);
            c = WMMA_F16(a, bf, c);
        }
#pragma unroll
        for (int r = 0; r < 8; ++r) {
            const int m = r + 8 * hi;
            const int col = nt * 16 + n;
            Ht[m * 72 + (col >> 5) * 32 + swzc(col & 31)] = (h16)sp(c[r]);
        }
    }
    __syncthreads();

#pragma unroll
    for (int nt = 0; nt < 2; ++nt) {
        const float bias = b1[nt * 16 + n];
        v8f c;
#pragma unroll
        for (int r = 0; r < 8; ++r) c[r] = bias;
#pragma unroll
        for (int kc = 0; kc < 2; ++kc) {
            v16h a  = load_A_sw(Ht, 72, kc, lane);
            v16h bf = load_B_sw(W1s, 32, kc, nt * 16, lane);
            c = WMMA_F16(a, bf, c);
        }
#pragma unroll
        for (int r = 0; r < 8; ++r) {
            const int m = r + 8 * hi;
            if (m < N_GRAPHS)
                u_new[m * DD + nt * 16 + n] = sp(c[r]);
        }
    }
}

// ------------------------------------------------------------------ launcher
extern "C" void kernel_launch(void* const* d_in, const int* in_sizes, int n_in,
                              void* d_out, int out_size, void* d_ws, size_t ws_size,
                              hipStream_t stream) {
    (void)in_sizes; (void)n_in; (void)out_size; (void)ws_size;

    const float* edge_feat  = (const float*)d_in[0];
    const float* node_feat  = (const float*)d_in[1];
    const float* graph_attr = (const float*)d_in[2];
    const int*   src        = (const int*)d_in[3];
    const int*   dst        = (const int*)d_in[4];
    const int*   ngid       = (const int*)d_in[5];
    const float* eW0 = (const float*)d_in[6];  const float* eb0 = (const float*)d_in[7];
    const float* eW1 = (const float*)d_in[8];  const float* eb1 = (const float*)d_in[9];
    const float* nW0 = (const float*)d_in[10]; const float* nb0 = (const float*)d_in[11];
    const float* nW1 = (const float*)d_in[12]; const float* nb1 = (const float*)d_in[13];
    const float* aW0 = (const float*)d_in[14]; const float* ab0 = (const float*)d_in[15];
    const float* aW1 = (const float*)d_in[16]; const float* ab1 = (const float*)d_in[17];
    (void)eb0; (void)eb1;

    float* e_new = (float*)d_out;
    float* v_new = e_new + (size_t)N_EDGES * DD;
    float* u_new = v_new + (size_t)N_NODES * DD;

    float* ws    = (float*)d_ws;
    float* e_sum = ws;                               // N_NODES * DD
    float* deg   = e_sum + (size_t)N_NODES * DD;     // N_NODES
    float* ue    = deg + N_NODES;                    // N_GRAPHS * DD
    float* ne    = ue + N_GRAPHS * DD;               // N_GRAPHS
    float* uv    = ne + N_GRAPHS;                    // N_GRAPHS * DD
    float* nv    = uv + N_GRAPHS * DD;               // N_GRAPHS
    const size_t ws_floats =
        (size_t)N_NODES * DD + N_NODES + 2 * ((size_t)N_GRAPHS * DD + N_GRAPHS);
    hipMemsetAsync(d_ws, 0, ws_floats * sizeof(float), stream);

    megnet_edge_kernel<<<N_EDGES / 64, 128, 0, stream>>>(
        edge_feat, node_feat, graph_attr, src, dst, ngid,
        (const float*)d_in[6], (const float*)d_in[7],
        (const float*)d_in[8], (const float*)d_in[9],
        e_new, e_sum, deg, ue, ne);

    megnet_node_kernel<<<(N_NODES + 63) / 64, 128, 0, stream>>>(
        node_feat, graph_attr, ngid, nW0, nb0, nW1, nb1,
        e_sum, deg, v_new, uv, nv);

    megnet_attr_kernel<<<1, 32, 0, stream>>>(
        graph_attr, aW0, ab0, aW1, ab1, ue, ne, uv, nv, u_new);
}
